// CrossModalGatedBottleneckAttention_1365799600459
// MI455X (gfx1250) — compile-verified, matrix operand order checked by reading, not run on hardware
//
#include <hip/hip_runtime.h>
#include <hip/hip_bf16.h>
#include <math.h>

typedef unsigned short u16;
typedef unsigned int   u32;

typedef __attribute__((ext_vector_type(16))) __bf16 v16bf;
typedef __attribute__((ext_vector_type(8)))  float  v8f;
typedef __attribute__((ext_vector_type(8)))  u32    v8u;
typedef __attribute__((ext_vector_type(4)))  u32    v4u;

union FragBF { v8u u; v16bf b; };

__device__ __forceinline__ float bf2f(u16 x) {
    u32 t = ((u32)x) << 16;
    return __uint_as_float(t);
}
__device__ __forceinline__ u16 f2bf(float f) {
    u32 b = __float_as_uint(f);
    b += 0x7FFFu + ((b >> 16) & 1u);   // round-to-nearest-even
    return (u16)(b >> 16);
}

// ---------------------------------------------------------------------------
// TDM: DMA a 2D tile (rows x rowlen bf16 elements, row stride in elements)
// from global memory into LDS at byte offset lds_off. Issued once per wave;
// completion tracked by TENSORcnt. D# layout per CDNA5 ISA.
// ---------------------------------------------------------------------------
__device__ __forceinline__ void tdm_load_2d_bf16(u32 lds_off, const u16* gptr,
                                                 int rows, int rowlen, int stride)
{
    unsigned long long ga = (unsigned long long)(uintptr_t)gptr;
    v4u g0;
    g0[0] = 1u;                               // count=1, user mode, no gather
    g0[1] = lds_off;                          // lds_addr (bytes)
    g0[2] = (u32)ga;                          // global_addr[31:0]
    g0[3] = ((u32)(ga >> 32) & 0x01FFFFFFu)   // global_addr[56:32]
            | 0x80000000u;                    // type=2 ("image")
    v8u g1;
    g1[0] = 0x00010000u;                      // wg_mask=0, data_size=1 (2B)
    g1[1] = 0xFFFF0000u;                      // tensor_dim0 lo16 (huge: no OOB clamp)
    g1[2] = 0xFFFF7FFFu;                      // tensor_dim0 hi16, tensor_dim1 lo16
    g1[3] = ((u32)rowlen << 16) | 0x7FFFu;    // tensor_dim1 hi16 | tile_dim0
    g1[4] = (u32)rows;                        // tile_dim1 (tile_dim2=0)
    g1[5] = (u32)stride;                      // tensor_dim0_stride[31:0]
    g1[6] = 0u;                               // stride[47:32] | dim1_stride lo
    g1[7] = 0u;
    asm volatile("tensor_load_to_lds %0, %1" :: "s"(g0), "s"(g1) : "memory");
}

// ---------------------------------------------------------------------------
// Generic batched bf16 WMMA GEMM:  C = alpha * A x op(B) [+ C] [+ bias]
//   A: [M,K] bf16 row-major (lda), B: [K,N] (ldb) or transB -> [N,K] (ldb)
//   Batch z: b1 = z / nB2, b2 = z % nB2; operand offset b1*s?1 + b2*s?2.
//   Block tile 128x64, K-stage 64 (2 WMMA K-slices), 8 wave32 waves (4x2),
//   wave tile 32x32 -> 8 v_wmma_f32_16x16x32_bf16 per stage.
//   A tile (and B tile in NT mode) staged via tensor_load_to_lds (TDM).
// ---------------------------------------------------------------------------
#define TM 128
#define TN 64
#define KT 64

__global__ __launch_bounds__(256)
void gemm_bf16_wmma(const u16* __restrict__ A, const u16* __restrict__ Bm,
                    void* __restrict__ C,
                    int M, int Nn, int K, int lda, int ldb, int ldc,
                    long sA1, long sB1, long sC1, int nB2,
                    long sA2, long sB2, long sC2,
                    float alpha, int transB, int accum,
                    const float* __restrict__ bias, int outf32)
{
    __shared__ u16 As[TM][KT];   // 16 KB
    __shared__ u16 Bs[TN][KT];   // 8 KB, stored [n][k] (matches A frag layout)

    const int z  = blockIdx.z;
    const int b1 = z / nB2;
    const int b2 = z - b1 * nB2;
    const u16* Ab = A  + (long)b1 * sA1 + (long)b2 * sA2;
    const u16* Bb = Bm + (long)b1 * sB1 + (long)b2 * sB2;
    const long coff = (long)b1 * sC1 + (long)b2 * sC2;

    const int tid  = threadIdx.x;
    const int lane = tid & 31;
    const int wave = tid >> 5;
    const int wm   = wave >> 1;          // 0..3
    const int wn   = wave & 1;           // 0..1
    const int m0   = blockIdx.x * TM;
    const int n0   = blockIdx.y * TN;

    const u32 ldsA = (u32)(uintptr_t)&As[0][0];   // low 32 bits == LDS byte addr
    const u32 ldsB = (u32)(uintptr_t)&Bs[0][0];

    v8f acc[2][2] = {};

    const int frow  = lane & 15;         // fragment row (A) / col (B)
    const int khalf = lane >> 4;         // K-half select per ISA layout

    for (int k0 = 0; k0 < K; k0 += KT) {
        // ---- stage A (always) and B (NT mode) via the Tensor Data Mover ----
        if (wave == 0) {
            tdm_load_2d_bf16(ldsA, Ab + (long)m0 * lda + k0, TM, KT, lda);
            if (transB)
                tdm_load_2d_bf16(ldsB, Bb + (long)n0 * ldb + k0, TN, KT, ldb);
        }
        if (!transB) {
            // B row-major [K,N]: 64x64 tile, transpose-scatter into Bs[n][k].
            // 512 x 16B chunks, 2 per thread; both loads in flight first.
            const int r0 = tid >> 3;          // k-row 0..31
            const int r1 = r0 + 32;           // k-row 32..63
            const int s  = tid & 7;           // n-group (8 cols)
            uint4 va = *(const uint4*)&Bb[(long)(k0 + r0) * ldb + n0 + s * 8];
            uint4 vb = *(const uint4*)&Bb[(long)(k0 + r1) * ldb + n0 + s * 8];
            const u16* pa = (const u16*)&va;
            const u16* pb = (const u16*)&vb;
            #pragma unroll
            for (int i = 0; i < 8; i++) Bs[s * 8 + i][r0] = pa[i];
            #pragma unroll
            for (int i = 0; i < 8; i++) Bs[s * 8 + i][r1] = pb[i];
        }
        if (wave == 0) __builtin_amdgcn_s_wait_tensorcnt(0);
        __syncthreads();

        // ---- fragment gathers + 8 WMMAs (2 M-tiles x 2 N-tiles x 2 K-slices)
        FragBF afr[2][2], bfr[2][2];
        #pragma unroll
        for (int t = 0; t < 2; t++) {
            #pragma unroll
            for (int i = 0; i < 2; i++) {
                const u16* src = &As[wm * 32 + i * 16 + frow][t * 32];
                #pragma unroll
                for (int p = 0; p < 8; p++) {
                    int kk = ((p < 4) ? 2 * p : 8 + 2 * p) + 8 * khalf;
                    afr[i][t].u[p] = *(const u32*)&src[kk];
                }
            }
            #pragma unroll
            for (int j = 0; j < 2; j++) {
                const u16* src = &Bs[wn * 32 + j * 16 + frow][t * 32];
                #pragma unroll
                for (int p = 0; p < 8; p++) {
                    int kk = ((p < 4) ? 2 * p : 8 + 2 * p) + 8 * khalf;
                    bfr[j][t].u[p] = *(const u32*)&src[kk];
                }
            }
        }
        #pragma unroll
        for (int t = 0; t < 2; t++)
            #pragma unroll
            for (int i = 0; i < 2; i++)
                #pragma unroll
                for (int j = 0; j < 2; j++)
                    acc[i][j] = __builtin_amdgcn_wmma_f32_16x16x32_bf16(
                        false, afr[i][t].b, false, bfr[j][t].b,
                        (short)0, acc[i][j], false, false);
        __syncthreads();
    }

    // ---- epilogue: C layout per ISA (VGPR r -> m = r + 8*(lane>=16), n = lane&15)
    const int mofs = khalf * 8;
    #pragma unroll
    for (int i = 0; i < 2; i++) {
        #pragma unroll
        for (int j = 0; j < 2; j++) {
            int ncol = n0 + wn * 32 + j * 16 + frow;
            float bv = bias ? bias[ncol] : 0.0f;
            #pragma unroll
            for (int r = 0; r < 8; r++) {
                long m   = (long)m0 + wm * 32 + i * 16 + mofs + r;
                long idx = coff + m * (long)ldc + ncol;
                float v  = alpha * acc[i][j][r] + bv;
                if (outf32) {
                    float* Cf = (float*)C;
                    if (accum) v += Cf[idx];
                    Cf[idx] = v;
                } else {
                    u16* Cb = (u16*)C;
                    if (accum) v += bf2f(Cb[idx]);
                    Cb[idx] = f2bf(v);
                }
            }
        }
    }
}

// ---------------------------------------------------------------------------
// Row softmax, in place on bf16. One block per row, L <= 768.
// ---------------------------------------------------------------------------
__global__ __launch_bounds__(256)
void softmax_inplace_bf16(u16* __restrict__ data, int L)
{
    long row = blockIdx.x;
    u16* p = data + row * (long)L;
    int t = threadIdx.x;
    __shared__ float red[256];

    float mx = -3.0e38f;
    for (int i = t; i < L; i += 256) mx = fmaxf(mx, bf2f(p[i]));
    red[t] = mx; __syncthreads();
    for (int s = 128; s > 0; s >>= 1) {
        if (t < s) red[t] = fmaxf(red[t], red[t + s]);
        __syncthreads();
    }
    mx = red[0]; __syncthreads();

    float vals[3];
    int cnt = 0;
    float sum = 0.0f;
    for (int i = t; i < L; i += 256) {
        float e = __expf(bf2f(p[i]) - mx);
        vals[cnt++] = e;
        sum += e;
    }
    red[t] = sum; __syncthreads();
    for (int s = 128; s > 0; s >>= 1) {
        if (t < s) red[t] += red[t + s];
        __syncthreads();
    }
    float inv = 1.0f / red[0];
    cnt = 0;
    for (int i = t; i < L; i += 256) p[i] = f2bf(vals[cnt++] * inv);
}

// ---------------------------------------------------------------------------
// Elementwise helpers
// ---------------------------------------------------------------------------
__global__ void f32_to_bf16_k(const float* __restrict__ in, u16* __restrict__ out,
                              long n, float scale)
{
    long i = blockIdx.x * (long)blockDim.x + threadIdx.x;
    long st = (long)gridDim.x * blockDim.x;
    for (; i < n; i += st) out[i] = f2bf(in[i] * scale);
}

__global__ void add_half_k(const u16* __restrict__ a, const u16* __restrict__ b,
                           u16* __restrict__ out, long n)
{
    long i = blockIdx.x * (long)blockDim.x + threadIdx.x;
    long st = (long)gridDim.x * blockDim.x;
    for (; i < n; i += st) out[i] = f2bf((bf2f(a[i]) + bf2f(b[i])) * 0.5f);
}

// h = relu(x1 + m_ij * sigmoid(f_logits))   (biases already folded by GEMM)
__global__ void gate_fuse_k(const float* __restrict__ x1,
                            const u16* __restrict__ mij,
                            const u16* __restrict__ flog,
                            u16* __restrict__ h, long n)
{
    long i = blockIdx.x * (long)blockDim.x + threadIdx.x;
    long st = (long)gridDim.x * blockDim.x;
    for (; i < n; i += st) {
        float f = 1.0f / (1.0f + __expf(-bf2f(flog[i])));
        float v = x1[i] + bf2f(mij[i]) * f;
        h[i] = f2bf(fmaxf(v, 0.0f));
    }
}

// ---------------------------------------------------------------------------
// Host side
// ---------------------------------------------------------------------------
static inline void launch_gemm(hipStream_t st, const u16* A, const u16* B, void* C,
                               int M, int N, int K, int lda, int ldb, int ldc,
                               long sA1, long sB1, long sC1, int nBatch, int nB2,
                               long sA2, long sB2, long sC2,
                               float alpha, int transB, int accum,
                               const float* bias, int outf32)
{
    dim3 g(M / TM, N / TN, nBatch);
    hipLaunchKernelGGL(gemm_bf16_wmma, g, dim3(256), 0, st,
                       A, B, C, M, N, K, lda, ldb, ldc,
                       sA1, sB1, sC1, nB2, sA2, sB2, sC2,
                       alpha, transB, accum, bias, outf32);
}

extern "C" void kernel_launch(void* const* d_in, const int* in_sizes, int n_in,
                              void* d_out, int out_size, void* d_ws, size_t ws_size,
                              hipStream_t stream)
{
    enum { Bn = 16, Nq = 768, Ff = 512, Nb = 256, Hh = 8, Hd = 64, F3 = 1536 };
    const long BNF = (long)Bn * Nq * Ff;

    const float* x1     = (const float*)d_in[0];
    const float* x2     = (const float*)d_in[1];
    const float* z_b    = (const float*)d_in[2];
    const float* Wqkv_i = (const float*)d_in[3];
    const float* Wqkv_j = (const float*)d_in[4];
    const float* Wqkv_b = (const float*)d_in[5];
    const float* W_f    = (const float*)d_in[6];
    const float* b_f    = (const float*)d_in[7];
    const float* W_m    = (const float*)d_in[8];
    const float* b_m    = (const float*)d_in[9];
    const float* W_QKV  = (const float*)d_in[10];
    const float* W_proj = (const float*)d_in[11];
    const float* b_proj = (const float*)d_in[12];
    float* out = (float*)d_out;

    // bump allocator over workspace
    char* wsb = (char*)d_ws;
    size_t off = 0;
    auto alloc16 = [&](long nelem) -> u16* {
        u16* p = (u16*)(wsb + off);
        off = (off + (size_t)nelem * 2 + 255) & ~(size_t)255;
        return p;
    };

    u16* x1b  = alloc16(BNF);
    u16* x2b  = alloc16(BNF);
    u16* zbb  = alloc16((long)Nb * Ff);
    u16* Wib  = alloc16((long)Ff * F3);
    u16* Wjb  = alloc16((long)Ff * F3);
    u16* Wbb  = alloc16((long)Ff * F3);
    u16* Wfb  = alloc16((long)2 * Ff * Ff);
    u16* Wmb  = alloc16((long)Ff * Ff);
    u16* WQb  = alloc16((long)Ff * F3);
    u16* Wpb  = alloc16((long)Ff * Ff);
    u16* k_i  = alloc16(BNF);
    u16* v_i  = alloc16(BNF);
    u16* q_j  = alloc16(BNF);
    u16* qkvb = alloc16((long)Nb * F3);
    u16* s_ib = alloc16((long)Bn * Nb * Nq);
    u16* a_ib = alloc16((long)Bn * Nb * Ff);
    u16* s_bj = alloc16((long)Bn * Nq * Nb);
    u16* a_bj = alloc16(BNF);
    u16* s_up = alloc16((long)Bn * Nq * Nb);
    u16* aib2 = alloc16(BNF);
    u16* aijb = alloc16(BNF);
    u16* flog = alloc16(BNF);
    u16* mij  = alloc16(BNF);
    u16* hb   = alloc16(BNF);
    u16* QKV  = alloc16((long)Bn * Nq * F3);
    u16* Ssc  = alloc16((long)Bn * Hh * Nq * Nq);
    u16* Hob  = alloc16(BNF);
    (void)ws_size; (void)in_sizes; (void)n_in; (void)out_size;

    const float scale = 1.0f / sqrtf((float)Ff);   // full-dim attention scale
    const float Scale = 1.0f / sqrtf((float)Hd);   // per-head scale

    dim3 eg(4096), eb(256);
    // ---- convert inputs & weights to bf16 (z_b pre-scaled by 0.2) ----
    hipLaunchKernelGGL(f32_to_bf16_k, eg, eb, 0, stream, x1, x1b, BNF, 1.0f);
    hipLaunchKernelGGL(f32_to_bf16_k, eg, eb, 0, stream, x2, x2b, BNF, 1.0f);
    hipLaunchKernelGGL(f32_to_bf16_k, eg, eb, 0, stream, z_b, zbb, (long)Nb * Ff, 0.2f);
    hipLaunchKernelGGL(f32_to_bf16_k, eg, eb, 0, stream, Wqkv_i, Wib, (long)Ff * F3, 1.0f);
    hipLaunchKernelGGL(f32_to_bf16_k, eg, eb, 0, stream, Wqkv_j, Wjb, (long)Ff * F3, 1.0f);
    hipLaunchKernelGGL(f32_to_bf16_k, eg, eb, 0, stream, Wqkv_b, Wbb, (long)Ff * F3, 1.0f);
    hipLaunchKernelGGL(f32_to_bf16_k, eg, eb, 0, stream, W_f, Wfb, (long)2 * Ff * Ff, 1.0f);
    hipLaunchKernelGGL(f32_to_bf16_k, eg, eb, 0, stream, W_m, Wmb, (long)Ff * Ff, 1.0f);
    hipLaunchKernelGGL(f32_to_bf16_k, eg, eb, 0, stream, W_QKV, WQb, (long)Ff * F3, 1.0f);
    hipLaunchKernelGGL(f32_to_bf16_k, eg, eb, 0, stream, W_proj, Wpb, (long)Ff * Ff, 1.0f);

    // ---- projections ----
    launch_gemm(stream, x1b, Wib + Ff,     k_i, Bn * Nq, Ff, Ff, Ff, F3, Ff,
                0, 0, 0, 1, 1, 0, 0, 0, 1.0f, 0, 0, nullptr, 0);
    launch_gemm(stream, x1b, Wib + 2 * Ff, v_i, Bn * Nq, Ff, Ff, Ff, F3, Ff,
                0, 0, 0, 1, 1, 0, 0, 0, 1.0f, 0, 0, nullptr, 0);
    launch_gemm(stream, x2b, Wjb,          q_j, Bn * Nq, Ff, Ff, Ff, F3, Ff,
                0, 0, 0, 1, 1, 0, 0, 0, 1.0f, 0, 0, nullptr, 0);
    launch_gemm(stream, zbb, Wbb, qkvb, Nb, F3, Ff, Ff, F3, F3,
                0, 0, 0, 1, 1, 0, 0, 0, 1.0f, 0, 0, nullptr, 0);

    // ---- a_ib = softmax(q_b k_i^T * scale) v_i ----
    launch_gemm(stream, qkvb, k_i, s_ib, Nb, Nq, Ff, F3, Ff, Nq,
                0, (long)Nq * Ff, (long)Nb * Nq, Bn, 1, 0, 0, 0,
                scale, 1, 0, nullptr, 0);
    hipLaunchKernelGGL(softmax_inplace_bf16, dim3(Bn * Nb), dim3(256), 0, stream, s_ib, Nq);
    launch_gemm(stream, s_ib, v_i, a_ib, Nb, Ff, Nq, Nq, Ff, Ff,
                (long)Nb * Nq, (long)Nq * Ff, (long)Nb * Ff, Bn, 1, 0, 0, 0,
                1.0f, 0, 0, nullptr, 0);

    // ---- a_bj = softmax(q_j k_b^T * scale) v_b ----
    launch_gemm(stream, q_j, qkvb + Ff, s_bj, Nq, Nb, Ff, Ff, F3, Nb,
                (long)Nq * Ff, 0, (long)Nq * Nb, Bn, 1, 0, 0, 0,
                scale, 1, 0, nullptr, 0);
    hipLaunchKernelGGL(softmax_inplace_bf16, dim3(Bn * Nq), dim3(256), 0, stream, s_bj, Nb);
    launch_gemm(stream, s_bj, qkvb + 2 * Ff, a_bj, Nq, Ff, Nb, Nb, F3, Ff,
                (long)Nq * Nb, 0, (long)Nq * Ff, Bn, 1, 0, 0, 0,
                1.0f, 0, 0, nullptr, 0);

    // ---- attn_up = softmax(z_i a_ib^T * scale); a_ib2 = attn_up @ a_ib ----
    launch_gemm(stream, x1b, a_ib, s_up, Nq, Nb, Ff, Ff, Ff, Nb,
                (long)Nq * Ff, (long)Nb * Ff, (long)Nq * Nb, Bn, 1, 0, 0, 0,
                scale, 1, 0, nullptr, 0);
    hipLaunchKernelGGL(softmax_inplace_bf16, dim3(Bn * Nq), dim3(256), 0, stream, s_up, Nb);
    launch_gemm(stream, s_up, a_ib, aib2, Nq, Ff, Nb, Nb, Ff, Ff,
                (long)Nq * Nb, (long)Nb * Ff, (long)Nq * Ff, Bn, 1, 0, 0, 0,
                1.0f, 0, 0, nullptr, 0);

    // ---- a_ij = (a_ib2 + a_bj) * 0.5 ----
    hipLaunchKernelGGL(add_half_k, eg, eb, 0, stream, aib2, a_bj, aijb, BNF);

    // ---- gated fusion ----
    launch_gemm(stream, aijb, Wfb, flog, Bn * Nq, Ff, Ff, Ff, Ff, Ff,
                0, 0, 0, 1, 1, 0, 0, 0, 1.0f, 0, 0, nullptr, 0);
    launch_gemm(stream, x2b, Wfb + (long)Ff * Ff, flog, Bn * Nq, Ff, Ff, Ff, Ff, Ff,
                0, 0, 0, 1, 1, 0, 0, 0, 1.0f, 0, 1, b_f, 0);
    launch_gemm(stream, aijb, Wmb, mij, Bn * Nq, Ff, Ff, Ff, Ff, Ff,
                0, 0, 0, 1, 1, 0, 0, 0, 1.0f, 0, 0, b_m, 0);
    hipLaunchKernelGGL(gate_fuse_k, eg, eb, 0, stream, x1, mij, flog, hb, BNF);

    // ---- MHSA ----
    launch_gemm(stream, hb, WQb, QKV, Bn * Nq, F3, Ff, Ff, F3, F3,
                0, 0, 0, 1, 1, 0, 0, 0, 1.0f, 0, 0, nullptr, 0);
    launch_gemm(stream, QKV, QKV + Ff, Ssc, Nq, Nq, Hd, F3, F3, Nq,
                (long)Nq * F3, (long)Nq * F3, (long)Hh * Nq * Nq,
                Bn * Hh, Hh, Hd, Hd, (long)Nq * Nq,
                Scale, 1, 0, nullptr, 0);
    hipLaunchKernelGGL(softmax_inplace_bf16, dim3(Bn * Hh * Nq), dim3(256), 0, stream, Ssc, Nq);
    launch_gemm(stream, Ssc, QKV + 2 * Ff, Hob, Nq, Hd, Nq, Nq, F3, Ff,
                (long)Hh * Nq * Nq, (long)Nq * F3, (long)Nq * Ff,
                Bn * Hh, Hh, (long)Nq * Nq, Hd, Hd,
                1.0f, 0, 0, nullptr, 0);

    // ---- out = H @ W_proj + b_proj (f32) ----
    launch_gemm(stream, Hob, Wpb, out, Bn * Nq, Ff, Ff, Ff, Ff, Ff,
                0, 0, 0, 1, 1, 0, 0, 0, 1.0f, 0, 0, b_proj, 1);
}
